// Encoder_5136780886514
// MI455X (gfx1250) — matrix operand order repaired and checked by reading, not compiled, not run on hardware
//
#include <hip/hip_runtime.h>

// ---------------------------------------------------------------------------
// Types for CDNA5 WMMA (wave32): v_wmma_f32_16x16x32_bf16
// ---------------------------------------------------------------------------
typedef __bf16 bf16x8  __attribute__((ext_vector_type(8)));
typedef __bf16 bf16x16 __attribute__((ext_vector_type(16)));
typedef float  f32x8   __attribute__((ext_vector_type(8)));

union BF16Frag { bf16x16 v; bf16x8 h[2]; };

// A-matrix 16x32 bf16 fragment (ISA 7.12.2): lane<16 -> row=lane, K={kb..kb+7, kb+16..kb+23};
// lane>=16 -> row=lane-16, K offset +8. Two contiguous 16B loads per lane.
__device__ __forceinline__ bf16x16 load_a_frag(const __bf16* rowptr, int kb, int lane) {
    const int koff = (lane < 16) ? 0 : 8;
    BF16Frag f;
    f.h[0] = *(const bf16x8*)(rowptr + kb + koff);
    f.h[1] = *(const bf16x8*)(rowptr + kb + koff + 16);
    return f.v;
}

// B-matrix 32x16 bf16 fragment from transposed storage Bt[col][K]:
// lane<16 -> col=lane, K={kb..kb+7, kb+16..kb+23} ; lane>=16 -> col=lane-16, K offset +8.
__device__ __forceinline__ bf16x16 load_b_frag(const __bf16* colptr, int kb, int lane) {
    const int koff = (lane < 16) ? 0 : 8;
    BF16Frag f;
    f.h[0] = *(const bf16x8*)(colptr + kb + koff);
    f.h[1] = *(const bf16x8*)(colptr + kb + koff + 16);
    return f.v;
}

__device__ __forceinline__ f32x8 wmma_bf16(bf16x16 a, bf16x16 b, f32x8 c) {
    return __builtin_amdgcn_wmma_f32_16x16x32_bf16(
        /*neg_a=*/false, a, /*neg_b=*/false, b,
        /*c_mod=*/(short)0, c, /*reuse_a=*/false, /*reuse_b=*/false);
}

__device__ __forceinline__ float sigf(float x) { return 1.0f / (1.0f + __expf(-x)); }

// Problem constants
#define VOCABC 32000
#define EMD    512
#define HU     1024
#define G4     4096
#define BATCHC 64
#define SEQC   256

// ---------------------------------------------------------------------------
// Prep kernels
// ---------------------------------------------------------------------------
__global__ void f32_to_bf16_kernel(const float* __restrict__ src, __bf16* __restrict__ dst,
                                   unsigned long long n) {
    unsigned long long i = (unsigned long long)blockIdx.x * blockDim.x + threadIdx.x;
    if (i < n) dst[i] = (__bf16)src[i];
}

// src [K][N] f32 (row-major) -> dst [N][K] bf16
__global__ void transpose_bf16_kernel(const float* __restrict__ src, __bf16* __restrict__ dst,
                                      int K, int N) {
    unsigned long long i = (unsigned long long)blockIdx.x * blockDim.x + threadIdx.x;
    unsigned long long total = (unsigned long long)K * N;
    if (i < total) {
        int n = (int)(i / K);
        int k = (int)(i % K);
        dst[i] = (__bf16)src[(unsigned long long)k * N + n];
    }
}

__global__ void init_state_kernel(const float* __restrict__ h0, const float* __restrict__ c0,
                                  __bf16* __restrict__ hA, float* __restrict__ c, int n) {
    int i = blockIdx.x * blockDim.x + threadIdx.x;
    if (i < n) { hA[i] = (__bf16)h0[i]; c[i] = c0[i]; }
}

// ---------------------------------------------------------------------------
// Ping-pong double-buffered 32(M) x 64(N) wave-tile GEMM core, unrolled by 2:
// buffer slots [0]/[1] alternate roles inside the loop body -> no register
// rotation copies; pointers advance 64 elems/iter so loads use constant
// instruction offsets (0 / 32 / 64).
// ---------------------------------------------------------------------------
template <int K>
__device__ __forceinline__ void gemm_32x64(const __bf16* arow0, const __bf16* arow1,
                                           const __bf16* c0p, const __bf16* c1p,
                                           const __bf16* c2p, const __bf16* c3p,
                                           int lane, f32x8 acc[2][4]) {
    bf16x16 A0[2], A1[2], B0[2], B1[2], B2[2], B3[2];
    A0[0] = load_a_frag(arow0, 0, lane);
    A1[0] = load_a_frag(arow1, 0, lane);
    B0[0] = load_b_frag(c0p, 0, lane);
    B1[0] = load_b_frag(c1p, 0, lane);
    B2[0] = load_b_frag(c2p, 0, lane);
    B3[0] = load_b_frag(c3p, 0, lane);

#pragma unroll 1
    for (int i = 0; i < (K - 64) / 64; ++i) {
        __builtin_prefetch(arow0 + 128, 0, 3);
        __builtin_prefetch(arow1 + 128, 0, 3);
        // stage 1: load k+32 into slot1, compute k with slot0
        A0[1] = load_a_frag(arow0, 32, lane);
        A1[1] = load_a_frag(arow1, 32, lane);
        B0[1] = load_b_frag(c0p, 32, lane);
        B1[1] = load_b_frag(c1p, 32, lane);
        B2[1] = load_b_frag(c2p, 32, lane);
        B3[1] = load_b_frag(c3p, 32, lane);
        acc[0][0] = wmma_bf16(A0[0], B0[0], acc[0][0]);
        acc[1][0] = wmma_bf16(A1[0], B0[0], acc[1][0]);
        acc[0][1] = wmma_bf16(A0[0], B1[0], acc[0][1]);
        acc[1][1] = wmma_bf16(A1[0], B1[0], acc[1][1]);
        acc[0][2] = wmma_bf16(A0[0], B2[0], acc[0][2]);
        acc[1][2] = wmma_bf16(A1[0], B2[0], acc[1][2]);
        acc[0][3] = wmma_bf16(A0[0], B3[0], acc[0][3]);
        acc[1][3] = wmma_bf16(A1[0], B3[0], acc[1][3]);
        // stage 2: load k+64 into slot0, compute k+32 with slot1
        A0[0] = load_a_frag(arow0, 64, lane);
        A1[0] = load_a_frag(arow1, 64, lane);
        B0[0] = load_b_frag(c0p, 64, lane);
        B1[0] = load_b_frag(c1p, 64, lane);
        B2[0] = load_b_frag(c2p, 64, lane);
        B3[0] = load_b_frag(c3p, 64, lane);
        acc[0][0] = wmma_bf16(A0[1], B0[1], acc[0][0]);
        acc[1][0] = wmma_bf16(A1[1], B0[1], acc[1][0]);
        acc[0][1] = wmma_bf16(A0[1], B1[1], acc[0][1]);
        acc[1][1] = wmma_bf16(A1[1], B1[1], acc[1][1]);
        acc[0][2] = wmma_bf16(A0[1], B2[1], acc[0][2]);
        acc[1][2] = wmma_bf16(A1[1], B2[1], acc[1][2]);
        acc[0][3] = wmma_bf16(A0[1], B3[1], acc[0][3]);
        acc[1][3] = wmma_bf16(A1[1], B3[1], acc[1][3]);
        arow0 += 64; arow1 += 64;
        c0p += 64; c1p += 64; c2p += 64; c3p += 64;
    }
    // tail: slot0 holds block K-64; load final block K-32 into slot1
    A0[1] = load_a_frag(arow0, 32, lane);
    A1[1] = load_a_frag(arow1, 32, lane);
    B0[1] = load_b_frag(c0p, 32, lane);
    B1[1] = load_b_frag(c1p, 32, lane);
    B2[1] = load_b_frag(c2p, 32, lane);
    B3[1] = load_b_frag(c3p, 32, lane);
    acc[0][0] = wmma_bf16(A0[0], B0[0], acc[0][0]);
    acc[1][0] = wmma_bf16(A1[0], B0[0], acc[1][0]);
    acc[0][1] = wmma_bf16(A0[0], B1[0], acc[0][1]);
    acc[1][1] = wmma_bf16(A1[0], B1[0], acc[1][1]);
    acc[0][2] = wmma_bf16(A0[0], B2[0], acc[0][2]);
    acc[1][2] = wmma_bf16(A1[0], B2[0], acc[1][2]);
    acc[0][3] = wmma_bf16(A0[0], B3[0], acc[0][3]);
    acc[1][3] = wmma_bf16(A1[0], B3[0], acc[1][3]);
    acc[0][0] = wmma_bf16(A0[1], B0[1], acc[0][0]);
    acc[1][0] = wmma_bf16(A1[1], B0[1], acc[1][0]);
    acc[0][1] = wmma_bf16(A0[1], B1[1], acc[0][1]);
    acc[1][1] = wmma_bf16(A1[1], B1[1], acc[1][1]);
    acc[0][2] = wmma_bf16(A0[1], B2[1], acc[0][2]);
    acc[1][2] = wmma_bf16(A1[1], B2[1], acc[1][2]);
    acc[0][3] = wmma_bf16(A0[1], B3[1], acc[0][3]);
    acc[1][3] = wmma_bf16(A1[1], B3[1], acc[1][3]);
}

// ---------------------------------------------------------------------------
// Phase 1: xW[t][b][4H] = emb[tokens[b][t]] @ W   (gather fused into GEMM)
// Block: 256 thr = 8 waves; block tile 32(M) x 512(N); wave tile 32 x 64.
// Row index m = t*64 + b -> xW row m directly.
// ---------------------------------------------------------------------------
__global__ void __launch_bounds__(256) xw_gemm_kernel(
    const int* __restrict__ tokens, const __bf16* __restrict__ embbf,
    const __bf16* __restrict__ Wt, float* __restrict__ xW) {
    const int lane = threadIdx.x & 31;
    const int wave = threadIdx.x >> 5;
    const int m0   = blockIdx.y * 32;
    const int n0   = blockIdx.x * 512 + wave * 64;
    const int r    = lane & 15;

    // A rows for both M-subtiles (embedding gather per lane)
    const int m0a = m0 + r, m1a = m0 + 16 + r;
    const int tok0 = tokens[(m0a & 63) * SEQC + (m0a >> 6)];
    const int tok1 = tokens[(m1a & 63) * SEQC + (m1a >> 6)];
    const __bf16* arow0 = embbf + (unsigned long long)tok0 * EMD;
    const __bf16* arow1 = embbf + (unsigned long long)tok1 * EMD;

    const __bf16* c0p = Wt + (unsigned long long)(n0 +  0 + r) * EMD;
    const __bf16* c1p = Wt + (unsigned long long)(n0 + 16 + r) * EMD;
    const __bf16* c2p = Wt + (unsigned long long)(n0 + 32 + r) * EMD;
    const __bf16* c3p = Wt + (unsigned long long)(n0 + 48 + r) * EMD;

    f32x8 acc[2][4] = {};
    gemm_32x64<EMD>(arow0, arow1, c0p, c1p, c2p, c3p, lane, acc);

    // C layout: VGPR v -> row base+v (lanes 0-15) / base+v+8 (lanes 16-31), col = n-tile + lane%16
    const int rowoff = (lane < 16) ? 0 : 8;
#pragma unroll
    for (int mt = 0; mt < 2; ++mt)
#pragma unroll
        for (int n = 0; n < 4; ++n)
#pragma unroll
            for (int v = 0; v < 8; ++v)
                xW[(unsigned long long)(m0 + mt * 16 + v + rowoff) * G4 +
                   (n0 + n * 16 + r)] = acc[mt][n][v];
}

// ---------------------------------------------------------------------------
// Phase 2: one LSTM step. Grid (8 unit-chunks x 2 batch-chunks), 256 thr.
// Block owns 32 batch rows x 128 units across ALL 4 gates:
//   wave w -> gate = w/2, 64-col half = (w%2). GEMM cols = gate*1024 + j0 + ...
// Wave accumulators staged through LDS, then fused gate epilogue.
// ---------------------------------------------------------------------------
__global__ void __launch_bounds__(256) lstm_step_kernel(
    const __bf16* __restrict__ hIn, const __bf16* __restrict__ Ut,
    const float* __restrict__ xW, const float* __restrict__ bias,
    float* __restrict__ c, __bf16* __restrict__ hOut,
    float* __restrict__ out, int t) {
    __shared__ float zs[4][32][128];   // 64 KB

    const int lane = threadIdx.x & 31;
    const int wave = threadIdx.x >> 5;
    const int mb   = blockIdx.y * 32;
    const int j0   = blockIdx.x * 128;
    const int gate = wave >> 1;
    const int nn   = (wave & 1) * 64;
    const int r    = lane & 15;

    const __bf16* arow0 = hIn + (unsigned long long)(mb + r) * HU;
    const __bf16* arow1 = arow0 + 16ull * HU;

    const unsigned long long cb = (unsigned long long)(gate * HU + j0 + nn + r) * HU;
    const __bf16* c0p = Ut + cb;
    const __bf16* c1p = Ut + cb + 16ull * HU;
    const __bf16* c2p = Ut + cb + 32ull * HU;
    const __bf16* c3p = Ut + cb + 48ull * HU;

    f32x8 acc[2][4] = {};
    gemm_32x64<HU>(arow0, arow1, c0p, c1p, c2p, c3p, lane, acc);

    const int rowoff = (lane < 16) ? 0 : 8;
#pragma unroll
    for (int mt = 0; mt < 2; ++mt)
#pragma unroll
        for (int n = 0; n < 4; ++n)
#pragma unroll
            for (int v = 0; v < 8; ++v)
                zs[gate][mt * 16 + v + rowoff][nn + n * 16 + r] = acc[mt][n][v];
    __syncthreads();

    const float* xWt = xW + (unsigned long long)t * BATCHC * G4;
    for (int e = threadIdx.x; e < 32 * 128; e += 256) {
        const int rr = e >> 7, u = e & 127;
        const int bi = mb + rr, unit = j0 + u;
        const unsigned long long xb = (unsigned long long)bi * G4 + unit;
        float zi = zs[0][rr][u] + xWt[xb         ] + bias[unit         ];
        float zf = zs[1][rr][u] + xWt[xb + 1*HU  ] + bias[unit + 1*HU  ];
        float zg = zs[2][rr][u] + xWt[xb + 2*HU  ] + bias[unit + 2*HU  ];
        float zo = zs[3][rr][u] + xWt[xb + 3*HU  ] + bias[unit + 3*HU  ];
        float ig = sigf(zi), fg = sigf(zf), gg = sigf(zg), og = sigf(zo);
        const unsigned long long ci = (unsigned long long)bi * HU + unit;
        float cv = fg * c[ci] + ig * gg;
        c[ci] = cv;
        float hv = og * sigf(cv);                    // output activation also sigmoid
        hOut[ci] = (__bf16)hv;
        out[((unsigned long long)bi * SEQC + t) * HU + unit] = hv;
    }
}

__global__ void finalize_kernel(const float* __restrict__ out, const float* __restrict__ c,
                                float* __restrict__ hT, float* __restrict__ cT) {
    int i = blockIdx.x * blockDim.x + threadIdx.x;
    if (i < BATCHC * HU) {
        int bi = i >> 10, u = i & (HU - 1);
        hT[i] = out[((unsigned long long)bi * SEQC + (SEQC - 1)) * HU + u];
        cT[i] = c[i];
    }
}

// ---------------------------------------------------------------------------
// Host side
// ---------------------------------------------------------------------------
extern "C" void kernel_launch(void* const* d_in, const int* in_sizes, int n_in,
                              void* d_out, int out_size, void* d_ws, size_t ws_size,
                              hipStream_t stream) {
    const int*   tokens = (const int*)d_in[0];
    const float* h0     = (const float*)d_in[1];
    const float* c0     = (const float*)d_in[2];
    const float* emb    = (const float*)d_in[3];
    const float* W      = (const float*)d_in[4];
    const float* U      = (const float*)d_in[5];
    const float* bias   = (const float*)d_in[6];
    float* out = (float*)d_out;

    char* ws = (char*)d_ws;
    size_t off = 0;
    auto alloc = [&](size_t bytes) -> void* {
        void* p = ws + off;
        off += (bytes + 255) & ~(size_t)255;
        return p;
    };
    float*  xW    = (float*)alloc((size_t)SEQC * BATCHC * G4 * sizeof(float));   // 256 MB
    __bf16* embbf = (__bf16*)alloc((size_t)VOCABC * EMD * sizeof(__bf16));       // 32 MB
    __bf16* Wt    = (__bf16*)alloc((size_t)G4 * EMD * sizeof(__bf16));           // 4 MB
    __bf16* Ut    = (__bf16*)alloc((size_t)G4 * HU * sizeof(__bf16));            // 8 MB
    float*  cbuf  = (float*)alloc((size_t)BATCHC * HU * sizeof(float));
    __bf16* hA    = (__bf16*)alloc((size_t)BATCHC * HU * sizeof(__bf16));
    __bf16* hB    = (__bf16*)alloc((size_t)BATCHC * HU * sizeof(__bf16));

    // Prep: bf16 embedding table; transposed bf16 weights
    {
        unsigned long long n = (unsigned long long)VOCABC * EMD;
        f32_to_bf16_kernel<<<(unsigned)((n + 255) / 256), 256, 0, stream>>>(emb, embbf, n);
    }
    {
        unsigned long long n = (unsigned long long)G4 * EMD;
        transpose_bf16_kernel<<<(unsigned)((n + 255) / 256), 256, 0, stream>>>(W, Wt, EMD, G4);
    }
    {
        unsigned long long n = (unsigned long long)G4 * HU;
        transpose_bf16_kernel<<<(unsigned)((n + 255) / 256), 256, 0, stream>>>(U, Ut, HU, G4);
    }
    init_state_kernel<<<(BATCHC * HU + 255) / 256, 256, 0, stream>>>(h0, c0, hA, cbuf,
                                                                    BATCHC * HU);

    // Phase 1: xW = gather(emb, tokens) @ W  -> [T][B][4H]
    {
        dim3 grid(G4 / 512, (SEQC * BATCHC) / 32);   // (8, 512)
        xw_gemm_kernel<<<grid, 256, 0, stream>>>(tokens, embbf, Wt, xW);
    }

    // Phase 2: 256 dependent steps, h ping-pongs between bf16 buffers
    {
        dim3 grid(HU / 128, BATCHC / 32);            // (8, 2)
        for (int t = 0; t < SEQC; ++t) {
            const __bf16* hin  = (t & 1) ? hB : hA;
            __bf16*       hout = (t & 1) ? hA : hB;
            lstm_step_kernel<<<grid, 256, 0, stream>>>(hin, Ut, xW, bias, cbuf, hout, out, t);
        }
    }

    // Tail: hT, cT appended after [B,T,H] output
    float* hT = out + (unsigned long long)BATCHC * SEQC * HU;
    float* cT = hT + (unsigned long long)BATCHC * HU;
    finalize_kernel<<<(BATCHC * HU + 255) / 256, 256, 0, stream>>>(out, cbuf, hT, cT);
}
